// LambdaConv_32916629357008
// MI455X (gfx1250) — matrix operand order repaired and checked by reading, not compiled, not run on hardware
//
#include <hip/hip_runtime.h>
#include <hip/hip_bf16.h>
#include <math.h>

// ---------------------------------------------------------------------------
// LambdaConv on MI455X (gfx1250, wave32).
//
// Shapes: B=16, C=256, W=H=64 (S=4096), heads=4, kh=4, u=1, vh=64, m=23, pad=11
//
// Strategy (memory-bound: ~270MB traffic vs ~10 GFLOP):
//  k0: prep  — build Wt[C=256][J=288] bf16 with BN scales folded into q/k rows;
//              qshift[16] = beta - mean*scale (k-shift cancels in softmax).
//  k1: WMMA  — fused q/k/v projection: OUT[s,j] = sum_c x[b,c,s]*Wt[c,j]
//              via v_wmma_f32_16x16x32_bf16; x read exactly once.
//  k2: softmax stats (max,sumexp) per (b, h*4+kc) over 4096 positions.
//  k3: Amap  — conv-then-spatial-sum collapses to per-position effective
//              weight A[h,k](x,y) = sum of in-bounds kernel taps.
//  k4: lambda[b,h,k,v] = sum_s (softmax(kk)+A) * v   (= lambda_c + lambda_p)
//  k5: out[b,h*64+vi,s] = sum_k q[b,s,h,k] * lambda[b,h,k,vi]
// ---------------------------------------------------------------------------

typedef __attribute__((ext_vector_type(16))) __bf16 v16bf;
typedef __attribute__((ext_vector_type(8)))  float  v8f;

union ABf { v16bf v; __bf16 e[16]; };

#define EPS 1e-5f

// ---------------------------------------------------------------- k0: prep
__global__ __launch_bounds__(256) void prep_kernel(
    const float* __restrict__ qw, const float* __restrict__ qg,
    const float* __restrict__ qb, const float* __restrict__ qm,
    const float* __restrict__ qv, const float* __restrict__ kw,
    const float* __restrict__ kg, const float* __restrict__ kv,
    const float* __restrict__ vw,
    __bf16* __restrict__ Wt, float* __restrict__ qshift)
{
    int idx = blockIdx.x * 256 + threadIdx.x;           // 288 blocks -> 73728
    if (idx < 256 * 288) {
        int c = idx / 288, j = idx % 288;
        float wvv;
        if (j < 16) {
            float sc = qg[j] * rsqrtf(qv[j] + EPS);
            wvv = qw[j * 256 + c] * sc;
        } else if (j < 32) {
            int jj = j - 16;
            float sc = kg[jj] * rsqrtf(kv[jj] + EPS);
            wvv = kw[jj * 256 + c] * sc;
        } else {
            wvv = vw[(j - 32) * 256 + c];
        }
        Wt[(size_t)c * 288 + j] = (__bf16)wvv;
    }
    if (idx < 16) {
        float sc = qg[idx] * rsqrtf(qv[idx] + EPS);
        qshift[idx] = qb[idx] - qm[idx] * sc;
    }
}

// ---------------------------------------------------- k1: fused projection
// grid = 512 blocks (16 b * 32 s-chunks), 256 threads = 8 waves, each wave
// owns a 16-position M-tile and all 288 output channels.
__global__ __launch_bounds__(256) void proj_wmma_kernel(
    const float* __restrict__ x, const __bf16* __restrict__ Wt,
    const float* __restrict__ qshift,
    float* __restrict__ qbuf, float* __restrict__ kkbuf,
    float* __restrict__ vbuf)
{
    const int tid  = threadIdx.x;
    const int lane = tid & 31;
    const int wid  = tid >> 5;
    const int l    = lane & 15;   // N index within tile / M index for A
    const int g    = lane >> 4;   // lane half
    const int blk  = blockIdx.x;
    const int b    = blk >> 5;
    const int s0   = (blk & 31) * 128 + wid * 16;

    // ---- A fragments: 16 rows (s) x 256 K (c), bf16, ISA 16-bit A layout:
    //      lane (g*16+l) elem i holds (M=l, K = (i<8 ? i : i+8) + g*8 + 32*kb)
    ABf A8[8];
    const float* xb = x + (size_t)b * 256 * 4096 + s0 + l;
#pragma unroll
    for (int kb = 0; kb < 8; ++kb) {
#pragma unroll
        for (int i = 0; i < 16; ++i) {
            int K = (i < 8 ? i : i + 8) + g * 8;
            A8[kb].e[i] = (__bf16)xb[(size_t)(kb * 32 + K) * 4096];
        }
    }
    const float qs = qshift[l];

    // ---- 18 N-tiles of 16 channels; 8 K-steps of 32 each
    for (int jt = 0; jt < 18; ++jt) {
        v8f acc = {0.f, 0.f, 0.f, 0.f, 0.f, 0.f, 0.f, 0.f};
        // B layout: lane (g*16+l) elem i holds (N=l, K = (i<8 ? i : i+8) + g*8)
        const __bf16* wp = Wt + (size_t)(g * 8) * 288 + jt * 16 + l;
#pragma unroll
        for (int kb = 0; kb < 8; ++kb) {
            ABf bv;
#pragma unroll
            for (int i = 0; i < 16; ++i)
                bv.e[i] = wp[(size_t)(kb * 32 + (i < 8 ? i : i + 8)) * 288];
            acc = __builtin_amdgcn_wmma_f32_16x16x32_bf16(
                false, A8[kb].v, false, bv.v, (short)0, acc, false, false);
        }
        // D layout: elem r on lane (g*16+l) is (M = r + g*8, N = l)
        const size_t rowbase = (size_t)b * 4096 + s0 + g * 8;
        if (jt == 0) {
#pragma unroll
            for (int r = 0; r < 8; ++r)
                qbuf[(rowbase + r) * 16 + l] = acc[r] + qs;   // BN shift
        } else if (jt == 1) {
#pragma unroll
            for (int r = 0; r < 8; ++r)
                kkbuf[(rowbase + r) * 16 + l] = acc[r];       // shift cancels
        } else {
#pragma unroll
            for (int r = 0; r < 8; ++r)
                vbuf[(rowbase + r) * 256 + (jt - 2) * 16 + l] = acc[r];
        }
    }
}

// ------------------------------------------------------- k2: softmax stats
__global__ __launch_bounds__(256) void stats_kernel(
    const float* __restrict__ kk, float* __restrict__ stats)
{
    __shared__ float red[256];
    const int bj = blockIdx.x;           // b*16 + j
    const int b = bj >> 4, j = bj & 15, t = threadIdx.x;
    const float* base = kk + (size_t)b * 4096 * 16 + j;

    float mx = -3.4e38f;
    for (int s = t; s < 4096; s += 256) mx = fmaxf(mx, base[(size_t)s * 16]);
    red[t] = mx; __syncthreads();
    for (int o = 128; o > 0; o >>= 1) {
        if (t < o) red[t] = fmaxf(red[t], red[t + o]);
        __syncthreads();
    }
    mx = red[0]; __syncthreads();

    float sm = 0.f;
    for (int s = t; s < 4096; s += 256) sm += __expf(base[(size_t)s * 16] - mx);
    red[t] = sm; __syncthreads();
    for (int o = 128; o > 0; o >>= 1) {
        if (t < o) red[t] += red[t + o];
        __syncthreads();
    }
    if (t == 0) { stats[bj * 2] = mx; stats[bj * 2 + 1] = red[0]; }
}

// ------------------------------------------- k3: effective positional weight
// A[h*4+k](x,y) = sum_{taps (i,j) with output in-bounds} emb[k][h][i][j]
__global__ __launch_bounds__(256) void amap_kernel(
    const float* __restrict__ emb, float* __restrict__ Amap)
{
    int idx = blockIdx.x * 256 + threadIdx.x;           // 256 blocks -> 65536
    int hk = idx >> 12, s = idx & 4095;
    int xs = s >> 6, ys = s & 63;
    int h = hk >> 2, kc = hk & 3;
    const float* wb = emb + (size_t)(kc * 4 + h) * 529; // [kh,heads,1,1,23,23]
    int ilo = max(0, xs - 52), ihi = min(22, xs + 11);
    int jlo = max(0, ys - 52), jhi = min(22, ys + 11);
    float acc = 0.f;
    for (int i = ilo; i <= ihi; ++i)
        for (int j = jlo; j <= jhi; ++j)
            acc += wb[i * 23 + j];
    Amap[(size_t)hk * 4096 + s] = acc;
}

// ------------------------------------- k4: lambda = sum_s (softmax+A) * v
// grid = 64 blocks (b,h); K=4096 reduction into [4 x 64]
__global__ __launch_bounds__(256) void lambda_kernel(
    const float* __restrict__ kk, const float* __restrict__ vbuf,
    const float* __restrict__ Amap, const float* __restrict__ stats,
    float* __restrict__ lambda)
{
    __shared__ float p[1024];
    const int bh = blockIdx.x;
    const int b = bh >> 2, h = bh & 3, t = threadIdx.x;
    float mx[4], inv[4];
#pragma unroll
    for (int k = 0; k < 4; ++k) {
        int bj = b * 16 + h * 4 + k;
        mx[k]  = stats[bj * 2];
        inv[k] = 1.f / stats[bj * 2 + 1];
    }
    const int vc = t & 63, sub = t >> 6;
    float acc0 = 0.f, acc1 = 0.f, acc2 = 0.f, acc3 = 0.f;

    for (int c = 0; c < 16; ++c) {
        const int s = c * 256 + t;
        float4 k4 = *(const float4*)(kk + ((size_t)b * 4096 + s) * 16 + h * 4);
        p[t * 4 + 0] = __expf(k4.x - mx[0]) * inv[0] + Amap[(h * 4 + 0) * 4096 + s];
        p[t * 4 + 1] = __expf(k4.y - mx[1]) * inv[1] + Amap[(h * 4 + 1) * 4096 + s];
        p[t * 4 + 2] = __expf(k4.z - mx[2]) * inv[2] + Amap[(h * 4 + 2) * 4096 + s];
        p[t * 4 + 3] = __expf(k4.w - mx[3]) * inv[3] + Amap[(h * 4 + 3) * 4096 + s];
        __syncthreads();
        const float* vp = vbuf + ((size_t)b * 4096 + c * 256 + sub * 64) * 256
                               + h * 64 + vc;
#pragma unroll 4
        for (int s2 = 0; s2 < 64; ++s2) {
            float vv = vp[(size_t)s2 * 256];
            int pb = (sub * 64 + s2) * 4;
            acc0 += p[pb + 0] * vv;
            acc1 += p[pb + 1] * vv;
            acc2 += p[pb + 2] * vv;
            acc3 += p[pb + 3] * vv;
        }
        __syncthreads();
    }

    float accs[4] = {acc0, acc1, acc2, acc3};
#pragma unroll
    for (int k = 0; k < 4; ++k) {
        __syncthreads();
        p[t] = accs[k];
        __syncthreads();
        if (t < 64) {
            float r = p[t] + p[64 + t] + p[128 + t] + p[192 + t];
            lambda[(((size_t)b * 4 + h) * 4 + k) * 64 + t] = r;
        }
    }
}

// ----------------------------------------------- k5: out = q . lambda
__global__ __launch_bounds__(256) void out_kernel(
    const float* __restrict__ qbuf, const float* __restrict__ lambda,
    float* __restrict__ out)
{
    __shared__ float lam[256];
    const int blk = blockIdx.x;               // 1024 = 16b * 4h * 16 s-chunks
    const int b = blk >> 6, rem = blk & 63;
    const int h = rem >> 4, sc = rem & 15;
    const int t = threadIdx.x;
    lam[t] = lambda[((size_t)b * 4 + h) * 256 + t];   // [k*64 + vi]
    __syncthreads();
    const int s = sc * 256 + t;
    float4 q4 = *(const float4*)(qbuf + ((size_t)b * 4096 + s) * 16 + h * 4);
    float* ob = out + ((size_t)b * 256 + h * 64) * 4096 + s;
#pragma unroll 8
    for (int vi = 0; vi < 64; ++vi) {
        float o = q4.x * lam[vi] + q4.y * lam[64 + vi]
                + q4.z * lam[128 + vi] + q4.w * lam[192 + vi];
        ob[(size_t)vi * 4096] = o;
    }
}

// --------------------------------------------------------------------------
extern "C" void kernel_launch(void* const* d_in, const int* in_sizes, int n_in,
                              void* d_out, int out_size, void* d_ws, size_t ws_size,
                              hipStream_t stream)
{
    const float* x       = (const float*)d_in[0];
    const float* q_w     = (const float*)d_in[1];
    const float* q_gamma = (const float*)d_in[2];
    const float* q_beta  = (const float*)d_in[3];
    const float* q_mean  = (const float*)d_in[4];
    const float* q_var   = (const float*)d_in[5];
    const float* k_w     = (const float*)d_in[6];
    const float* k_gamma = (const float*)d_in[7];
    // d_in[8]=k_beta (cancels in softmax), d_in[9]=k_mean (zeros; scale-fold only needs var)
    const float* k_var   = (const float*)d_in[10];
    const float* v_w     = (const float*)d_in[11];
    const float* emb     = (const float*)d_in[12];
    float* out = (float*)d_out;

    char* ws = (char*)d_ws;
    __bf16* Wt     = (__bf16*)(ws + 0);          // 256*288*2   = 147456
    float*  qshift = (float*) (ws + 147456);     // 64 (pad 256)
    float*  qbuf   = (float*) (ws + 147712);     // 16*4096*16*4 = 4 MB
    float*  kkbuf  = (float*) (ws + 4342016);    // 4 MB
    float*  stats  = (float*) (ws + 8536320);    // 256*2*4
    float*  Amap   = (float*) (ws + 8538368);    // 16*4096*4 = 256 KB
    float*  lambda = (float*) (ws + 8800512);    // 16*4*4*64*4 = 64 KB
    float*  vbuf   = (float*) (ws + 8866048);    // 16*4096*256*4 = 64 MB

    prep_kernel<<<288, 256, 0, stream>>>(q_w, q_gamma, q_beta, q_mean, q_var,
                                         k_w, k_gamma, k_var, v_w, Wt, qshift);
    proj_wmma_kernel<<<512, 256, 0, stream>>>(x, Wt, qshift, qbuf, kkbuf, vbuf);
    stats_kernel<<<256, 256, 0, stream>>>(kkbuf, stats);
    amap_kernel<<<256, 256, 0, stream>>>(emb, Amap);
    lambda_kernel<<<64, 256, 0, stream>>>(kkbuf, vbuf, Amap, stats, lambda);
    out_kernel<<<1024, 256, 0, stream>>>(qbuf, lambda, out);
}
